// BasicBlock_25383256719869
// MI455X (gfx1250) — 8-bit matrix operand order repaired and checked by reading, compiled, not run on hardware
//
#include <hip/hip_runtime.h>

typedef __attribute__((ext_vector_type(8))) int v8i;

#define B_   32
#define CIN  128
#define COUT 128
#define H_   56
#define W_   56
#define HW   (H_ * W_)
#define HP   58
#define WP   58
#define KTOT (CIN * 9)
#define WBYTES (COUT * KTOT)
#define PADSLAB ((long)HP * WP * CIN)
#define TPW  4

__global__ __launch_bounds__(128) void prep_w(
    const float* __restrict__ Wt, const float* __restrict__ gamma,
    const float* __restrict__ beta, const float* __restrict__ rmean,
    const float* __restrict__ rvar, signed char* __restrict__ bw8,
    float* __restrict__ alpha, float* __restrict__ shiftv)
{
    const int co  = blockIdx.x;
    const int tid = threadIdx.x;
    __shared__ float red[128];

    float s = 0.f;
    for (int k = tid; k < KTOT; k += 128) s += fabsf(Wt[co * KTOT + k]);
    red[tid] = s;
    __syncthreads();
    #pragma unroll
    for (int off = 64; off > 0; off >>= 1) {
        if (tid < off) red[tid] += red[tid + off];
        __syncthreads();
    }
    if (tid == 0) {
        float scale = red[0] * (1.0f / (float)KTOT);
        float inv   = gamma[co] * rsqrtf(rvar[co] + 1e-5f);
        alpha[co]  = scale * inv;
        shiftv[co] = beta[co] - rmean[co] * inv;
    }

    for (int k = tid; k < KTOT; k += 128) {
        float v = Wt[co * KTOT + k];
        signed char sg = (v > 0.f) ? 1 : ((v < 0.f) ? -1 : 0);
        int c = k / 9, r = k % 9;
        bw8[co * KTOT + r * CIN + c] = sg;
    }
}

__global__ __launch_bounds__(256) void zero_halo(signed char* __restrict__ bxs)
{
    const int b   = blockIdx.x;
    const int tid = threadIdx.x;
    int4* slab = (int4*)(bxs + (long)b * PADSLAB);
    const int4 z = {0, 0, 0, 0};
    for (int i = tid; i < 1824; i += 256) {
        int off;
        if (i < 464) {
            off = i;
        } else if (i < 928) {
            off = 57 * WP * 8 + (i - 464);
        } else {
            int j    = i - 928;
            int side = j / 448;
            int jj   = j % 448;
            int pix  = jj / 8;
            int q    = jj % 8;
            off = ((pix + 1) * WP + side * 57) * 8 + q;
        }
        slab[off] = z;
    }
}

__global__ __launch_bounds__(256) void binarize_x(
    const float* __restrict__ x, signed char* __restrict__ bxs)
{
    __shared__ signed char s[CIN * W_];
    const int bh  = blockIdx.x;
    const int b   = bh / H_;
    const int h   = bh % H_;
    const int tid = threadIdx.x;

    const float* xrow = x + (long)b * CIN * HW + (long)h * W_;
    for (int i = tid; i < CIN * W_; i += 256) {
        int c = i / W_, w = i % W_;
        float v = xrow[(long)c * HW + w];
        s[c * W_ + w] = (v > 0.f) ? 1 : ((v < 0.f) ? -1 : 0);
    }
    __syncthreads();

    int* orow32 = (int*)(bxs + (((long)b * HP + h + 1) * WP + 1) * CIN);
    for (int i = tid; i < W_ * (CIN / 4); i += 256) {
        int w = i / (CIN / 4);
        int c = (i % (CIN / 4)) * 4;
        unsigned int v =  (unsigned char)s[(c + 0) * W_ + w]
                       | ((unsigned char)s[(c + 1) * W_ + w] << 8)
                       | ((unsigned char)s[(c + 2) * W_ + w] << 16)
                       | ((unsigned char)s[(c + 3) * W_ + w] << 24);
        orow32[(long)w * (CIN / 4) + (c >> 2)] = (int)v;
    }
}

__global__ __launch_bounds__(256) void bireal_wmma(
    const signed char* __restrict__ bxs,
    const signed char* __restrict__ bw8,
    const float* __restrict__ x,
    const float* __restrict__ alpha,
    const float* __restrict__ shiftv,
    float* __restrict__ out)
{
    extern __shared__ char smem[];

    const int tid = threadIdx.x;
    {
        int4*       sw  = (int4*)smem;
        const int4* gw4 = (const int4*)bw8;
        #pragma unroll
        for (int i = tid; i < WBYTES / 16; i += 256) sw[i] = gw4[i];
    }
    __syncthreads();

    const int wave    = tid >> 5;
    const int lane    = tid & 31;
    const int row     = lane & 15;
    const int hi_half = lane >> 4;
    const int n_lo    = lane & 15;

    float al[8], sh[8];
    #pragma unroll
    for (int t = 0; t < 8; ++t) {
        al[t] = alpha[t * 16 + n_lo];
        sh[t] = shiftv[t * 16 + n_lo];
    }

    const int tile0 = (blockIdx.x * 8 + wave) * TPW;

    #pragma unroll 1
    for (int tt = 0; tt < TPW; ++tt) {
        asm volatile("" ::: "memory");

        const int p0  = (tile0 + tt) * 16;
        const int p   = p0 + row;
        const int b   = p / HW;
        const int rem = p % HW;
        const int h   = rem / W_;
        const int w   = rem % W_;

        __builtin_prefetch(x + (long)b * CIN * HW + rem, 0, 3);

        const signed char* abase =
            bxs + (((long)b * HP + h + 1) * WP + (w + 1)) * CIN + hi_half * 8;

        const v8i vzero = {0, 0, 0, 0, 0, 0, 0, 0};
        v8i acc[8];
        #pragma unroll
        for (int t = 0; t < 8; ++t) acc[t] = vzero;

        #pragma unroll
        for (int r = 0; r < 9; ++r) {
            const int dh   = r / 3 - 1;
            const int dw   = r % 3 - 1;
            const int toff = (dh * WP + dw) * CIN;

            #pragma unroll
            for (int ch = 0; ch < 2; ++ch) {
                const int c0 = ch * 64;
                union { unsigned long long q[4]; v8i v; } au;
                #pragma unroll
                for (int j = 0; j < 4; ++j) {
                    au.q[j] = *(const unsigned long long*)(abase + toff + c0 + 8 * j + 16 * (j >> 1) + hi_half * 8);
                }
                const int ks = r * 2 + ch;
                const int kb = ks * 64 + hi_half * 16;
                #pragma unroll
                for (int t = 0; t < 8; ++t) {
                    const int n = t * 16 + n_lo;
                    const int4* bp = (const int4*)(smem + n * KTOT + kb);
                    union { int4 i4[2]; v8i v; } bu;
                    bu.i4[0] = bp[0];
                    bu.i4[1] = bp[2];
                    acc[t] = __builtin_amdgcn_wmma_i32_16x16x64_iu8(
                         true, au.v,  true, bu.v,
                        acc[t],  false,  false);
                }
            }
        }

        #pragma unroll
        for (int i = 0; i < 8; ++i) {
            const int  pp    = p0 + i + hi_half * 8;
            const int  bb    = pp / HW;
            const int  rr    = pp % HW;
            const long obase = (long)bb * COUT * HW + rr;
            #pragma unroll
            for (int t = 0; t < 8; ++t) {
                const long idx = obase + (long)(t * 16 + n_lo) * HW;
                out[idx] = (float)acc[t][i] * al[t] + sh[t] + x[idx];
            }
        }
    }
}

extern "C" void kernel_launch(void* const* d_in, const int* in_sizes, int n_in,
                              void* d_out, int out_size, void* d_ws, size_t ws_size,
                              hipStream_t stream) {
    const float* x     = (const float*)d_in[0];
    const float* Wt    = (const float*)d_in[1];
    const float* gamma = (const float*)d_in[2];
    const float* beta  = (const float*)d_in[3];
    const float* rmean = (const float*)d_in[4];
    const float* rvar  = (const float*)d_in[5];
    float* out = (float*)d_out;

    char* ws = (char*)d_ws;
    signed char* bw8   = (signed char*)ws;
    float*       alpha = (float*)(ws + WBYTES);
    float*       shv   = (float*)(ws + WBYTES + 512);
    signed char* bxs   = (signed char*)(ws + WBYTES + 1024);

    prep_w<<<COUT, 128, 0, stream>>>(Wt, gamma, beta, rmean, rvar, bw8, alpha, shv);
    zero_halo<<<B_, 256, 0, stream>>>(bxs);
    binarize_x<<<B_ * H_, 256, 0, stream>>>(x, bxs);

    const int mtiles = (B_ * HW) / 16;
    bireal_wmma<<<mtiles / (8 * TPW), 256, WBYTES, stream>>>(bxs, bw8, x, alpha, shv, out);
}
